// GTS_Forecasting_Module_51651276702063
// MI455X (gfx1250) — matrix operand order repaired and checked by reading, not compiled, not run on hardware
//
#include <hip/hip_runtime.h>

typedef __attribute__((ext_vector_type(16))) __bf16 v16bf;
typedef __attribute__((ext_vector_type(8)))  float  v8f;
typedef unsigned short u16;
typedef unsigned int   u32;

// Async global->LDS path (CDNA5): guarded so the build stays green if the
// builtin is absent / differently named on this toolchain.
#if defined(__HIP_DEVICE_COMPILE__) && \
    __has_builtin(__builtin_amdgcn_global_load_async_to_lds_b128) && \
    __has_builtin(__builtin_amdgcn_s_wait_asynccnt)
#define HAVE_ASYNC_LDS 1
#else
#define HAVE_ASYNC_LDS 0
#endif

#if HAVE_ASYNC_LDS
// Param type per hipcc diagnostic: '__attribute__((vector_size(16))) int __device__ *'
typedef int v4i_vs __attribute__((__vector_size__(16)));
typedef __attribute__((address_space(1))) v4i_vs* as1_v4i;
typedef __attribute__((address_space(3))) v4i_vs* as3_v4i;
__device__ __forceinline__ void async_ld_b128(const void* g, void* l) {
  __builtin_amdgcn_global_load_async_to_lds_b128(
      (as1_v4i)(v4i_vs*)const_cast<void*>(g),
      (as3_v4i)(v4i_vs*)l, 0, 0);
}
#endif

// ---------------------------------------------------------------- helpers
__device__ __forceinline__ u16 f2bf(float f) {           // RNE float -> bf16
  u32 u = __float_as_uint(f);
  u += 0x7fffu + ((u >> 16) & 1u);
  return (u16)(u >> 16);
}
__device__ __forceinline__ float bf2f(u16 h) {
  return __uint_as_float(((u32)h) << 16);
}

// ---------------------------------------------------------------- zero words
__global__ void k_zero(u32* __restrict__ p, int nwords) {
  int i = blockIdx.x * 256 + threadIdx.x;
  if (i < nwords) p[i] = 0u;
}

// ---------------------------------------------------------------- f32 -> bf16
__global__ void k_f2bf(const float* __restrict__ s, u16* __restrict__ d, int n) {
  int i = blockIdx.x * 256 + threadIdx.x;
  if (i < n) d[i] = f2bf(s[i]);
}

// ---------------------------------------------------------------- conv1 + relu + bn1 stats
// grid = (32 channels, 2240 chunks), block = 256.  4096 nodes * 140 out-len.
__launch_bounds__(256)
__global__ void k_conv1(const float* __restrict__ x, const float* __restrict__ w,
                        const float* __restrict__ b, float* __restrict__ out1,
                        float* __restrict__ stats, int tstep) {
  __shared__ float wl[10];
  __shared__ float red[256];
  const int c   = blockIdx.x;
  const int tid = threadIdx.x;
  const int pos = blockIdx.y * 256 + tid;          // < 573440 exactly
  const int n   = pos / 140;
  const int l   = pos - n * 140;
  if (tid < 10) wl[tid] = w[c * 10 + tid];
  __syncthreads();
  const float* xr = x + (size_t)n * 564 + tstep * 12 + 2 * l;
  float v = b[c];
#pragma unroll
  for (int k = 0; k < 10; k++) v = fmaf(wl[k], xr[k], v);
  v = fmaxf(v, 0.0f);
  out1[((size_t)n * 32 + c) * 140 + l] = v;
  red[tid] = v; __syncthreads();
  for (int s = 128; s > 0; s >>= 1) { if (tid < s) red[tid] += red[tid + s]; __syncthreads(); }
  if (tid == 0) atomicAdd(&stats[c], red[0]);
  __syncthreads();
  red[tid] = v * v; __syncthreads();
  for (int s = 128; s > 0; s >>= 1) { if (tid < s) red[tid] += red[tid + s]; __syncthreads(); }
  if (tid == 0) atomicAdd(&stats[32 + c], red[0]);
}

// ---------------------------------------------------------------- bn1 scale/shift params
__global__ void k_bnparams(const float* __restrict__ stats, const float* __restrict__ g1,
                           const float* __restrict__ be1, float* __restrict__ bnp) {
  int c = threadIdx.x;                              // 32 threads
  const float cnt = 4096.0f * 140.0f;
  float m   = stats[c] / cnt;
  float var = stats[32 + c] / cnt - m * m;
  float s   = g1[c] * rsqrtf(var + 1e-5f);
  bnp[c] = s; bnp[32 + c] = be1[c] - m * s;
}

// ---------------------------------------------------------------- im2col (bn1 applied) -> bf16 patches
// P[(n*66+l2)][c1*10+k] = bf16(bn1(out1[n][c1][2*l2+k]));  86.5M elems
__launch_bounds__(256)
__global__ void k_im2col(const float* __restrict__ out1, const float* __restrict__ bnp,
                         u16* __restrict__ P) {
  int i  = blockIdx.x * 256 + threadIdx.x;          // < 270336*320 exactly
  int m  = i / 320, kk = i - m * 320;
  int n  = m / 66,  l2 = m - n * 66;
  int c1 = kk / 10, k  = kk - c1 * 10;
  float v = fmaf(bnp[c1], out1[((size_t)n * 32 + c1) * 140 + 2 * l2 + k], bnp[32 + c1]);
  P[i] = f2bf(v);
}

// ---------------------------------------------------------------- conv2 weight transpose -> bf16
__global__ void k_w2t(const float* __restrict__ w2, u16* __restrict__ w2t) {
  int i = blockIdx.x * 256 + threadIdx.x;           // < 320*64 exactly (grid 80)
  int kk = i >> 6, c2 = i & 63;
  w2t[i] = f2bf(w2[c2 * 320 + kk]);
}

// ---------------------------------------------------------------- bn2 stats over relu'd Z
// grid = (64 channels, 1056 chunks); Z layout [m=n*66+l][c2]
__launch_bounds__(256)
__global__ void k_stats2(const float* __restrict__ Z, float* __restrict__ stats) {
  __shared__ float red[256];
  const int c2  = blockIdx.x;
  const int tid = threadIdx.x;
  const int m   = blockIdx.y * 256 + tid;           // < 270336 exactly
  float v = Z[(size_t)m * 64 + c2];
  red[tid] = v; __syncthreads();
  for (int s = 128; s > 0; s >>= 1) { if (tid < s) red[tid] += red[tid + s]; __syncthreads(); }
  if (tid == 0) atomicAdd(&stats[64 + c2], red[0]);
  __syncthreads();
  red[tid] = v * v; __syncthreads();
  for (int s = 128; s > 0; s >>= 1) { if (tid < s) red[tid] += red[tid + s]; __syncthreads(); }
  if (tid == 0) atomicAdd(&stats[128 + c2], red[0]);
}

// ---------------------------------------------------------------- bn2 + fcconv + relu + fc -> bf16 emb
// grid = 4096 nodes, block = 64.  Z layout [n*66+l][c2].
__launch_bounds__(64)
__global__ void k_embed_tail(const float* __restrict__ Z, const float* __restrict__ stats,
                             const float* __restrict__ g2, const float* __restrict__ be2,
                             const float* __restrict__ fcw, const float* __restrict__ fcb,
                             const float* __restrict__ fc_w, const float* __restrict__ fc_b,
                             u16* __restrict__ embs, int tstep) {
  __shared__ float z[66];
  __shared__ float sc2[64], sh2[64];
  const int n = blockIdx.x;
  const int t = threadIdx.x;
  {
    const float cnt = 4096.0f * 66.0f;
    float m   = stats[64 + t] / cnt;
    float var = stats[128 + t] / cnt - m * m;
    float s   = g2[t] * rsqrtf(var + 1e-5f);
    sc2[t] = s; sh2[t] = be2[t] - m * s;
  }
  __syncthreads();
  for (int l = t; l < 66; l += 64) {
    const float* zr = Z + ((size_t)n * 66 + l) * 64;
    float v = fcb[0];
#pragma unroll 8
    for (int c = 0; c < 64; c++)
      v = fmaf(fcw[c], fmaf(sc2[c], zr[c], sh2[c]), v);
    z[l] = fmaxf(v, 0.0f);
  }
  __syncthreads();
  float e = fc_b[t];
#pragma unroll 6
  for (int l = 0; l < 66; l++) e = fmaf(z[l], fc_w[l * 64 + t], e);
  embs[((size_t)tstep * 4096 + n) * 64 + t] = f2bf(e);
}

// ---------------------------------------------------------------- generic bf16 WMMA GEMM
// C[M,N] = alpha*A@B (+bias[n]) (-Xsub[m,n]) (relu);  fp32 and/or bf16 out.
// 256 thr = 8 waves; block tile 128x64; wave: 16 rows x 64 cols; BK=32.
// A tile row-major (padded stride 40), B tile stored TRANSPOSED (Bt[n][k],
// padded stride 40) so every lane's fragment is one contiguous 32B run ->
// ds_load_b128 instead of 16x ds_load_u16.
#define LDA_S 40
#define BT_S  40
__launch_bounds__(256)
__global__ void k_gemm_bf16(const u16* __restrict__ A, int lda,
                            const u16* __restrict__ B, int ldb,
                            int M, int N, int K,
                            float* __restrict__ Cf, u16* __restrict__ Cbf, int ldc,
                            const float* __restrict__ bias,
                            const u16* __restrict__ Xsub, int ldx,
                            float alpha, int actRelu) {
  __shared__ u16 As[128 * LDA_S];
  __shared__ u16 Bt[64 * BT_S];
  const int tid  = threadIdx.x;
  const int wave = tid >> 5;
  const int lane = tid & 31;
  const int m0   = blockIdx.x * 128;
  const int n0   = blockIdx.y * 64;
  const int wrow = wave << 4;
  const int lm   = lane & 15;
  const int hi   = lane >> 4;

  // staging coordinates (fixed per thread)
  const int arow = tid >> 1;
  const int acol = (tid & 1) << 4;              // 0 or 16
  const u16* asrc0 = A + (size_t)(m0 + arow) * lda + acol;
  u16* adst = &As[arow * LDA_S + acol];
  const int bidx = tid * 8;
  const int brow = bidx >> 6;                   // 0..31
  const int bcol = bidx & 63;
  const u16* bsrc0 = B + (size_t)brow * ldb + n0 + bcol;

  v8f acc[4];
#pragma unroll
  for (int i = 0; i < 4; i++)
#pragma unroll
    for (int j = 0; j < 8; j++) acc[i][j] = 0.0f;

  for (int k0 = 0; k0 < K; k0 += 32) {
    // ---- stage A tile (128x32): 2 x b128 per thread
    const u16* asrc = asrc0 + k0;
#if HAVE_ASYNC_LDS
    async_ld_b128(asrc, adst);
    async_ld_b128(asrc + 8, adst + 8);
#else
    ((uint4*)adst)[0] = ((const uint4*)asrc)[0];
    ((uint4*)adst)[1] = ((const uint4*)asrc)[1];
#endif
    // ---- stage B tile (32x64) transposed: coalesced 16B load, 8x b16 scatter
    {
      union { uint4 q; u16 h[8]; } bv;
      bv.q = *(const uint4*)(bsrc0 + (size_t)k0 * ldb);
#pragma unroll
      for (int i = 0; i < 8; i++) Bt[(bcol + i) * BT_S + brow] = bv.h[i];
    }
#if HAVE_ASYNC_LDS
    __builtin_amdgcn_s_wait_asynccnt(0);
#endif
    __syncthreads();

    // ---- A fragment: k(e) = e + 16*hi  -> two 16B runs
    union { v16bf v; u32 w[8]; } af;
    {
      const u16* ap = &As[(wrow + lm) * LDA_S + (hi << 4)];
#pragma unroll
      for (int p = 0; p < 4; p++) af.w[p] = ((const u32*)ap)[p];
      const u16* ap2 = ap + 8;
#pragma unroll
      for (int p = 0; p < 4; p++) af.w[4 + p] = ((const u32*)ap2)[p];
    }
#pragma unroll
    for (int nt = 0; nt < 4; nt++) {
      // ---- B fragment: b[e] = B[hi*16+e][nt*16+lm] -> one 32B run in Bt
      union { v16bf v; u32 w[8]; } bfr;
      const u16* bp = &Bt[(nt * 16 + lm) * BT_S + (hi << 4)];
#pragma unroll
      for (int p = 0; p < 8; p++) bfr.w[p] = ((const u32*)bp)[p];
      acc[nt] = __builtin_amdgcn_wmma_f32_16x16x32_bf16(
          false, af.v, false, bfr.v, (short)0, acc[nt], false, false);
    }
    __syncthreads();
  }

  // ---- epilogue: C/D layout -> VGPR v is row (v + 8*hi), col = nt*16 + lm
#pragma unroll
  for (int nt = 0; nt < 4; nt++) {
#pragma unroll
    for (int v = 0; v < 8; v++) {
      const int m = m0 + wrow + v + (hi << 3);
      const int n = n0 + nt * 16 + lm;
      float val = alpha * acc[nt][v];
      if (bias) val += bias[n];
      if (Xsub) val -= bf2f(Xsub[(size_t)m * ldx + n]);
      if (actRelu) val = fmaxf(val, 0.0f);
      if (Cf)  Cf[(size_t)m * ldc + n]  = val;
      if (Cbf) Cbf[(size_t)m * ldc + n] = f2bf(val);
    }
  }
}

// ---------------------------------------------------------------- pack [x | h] into G cols 0..127
__global__ void k_pack(const u16* __restrict__ x, const u16* __restrict__ h, u16* __restrict__ G) {
  int i = blockIdx.x * 256 + threadIdx.x;   // < 4096*64
  int n = i >> 6, j = i & 63;
  G[(size_t)n * 384 + j]      = x[i];
  G[(size_t)n * 384 + 64 + j] = h[i];
}

// ---------------------------------------------------------------- r,u = sigmoid(ru); G[:,64:128] = bf16(r*h)
__global__ void k_gate(const float* __restrict__ ru, const float* __restrict__ h,
                       float* __restrict__ u, u16* __restrict__ G) {
  int i = blockIdx.x * 256 + threadIdx.x;   // < 4096*64
  int n = i >> 6, j = i & 63;
  float r  = 1.0f / (1.0f + __expf(-ru[(size_t)n * 128 + j]));
  float uu = 1.0f / (1.0f + __expf(-ru[(size_t)n * 128 + 64 + j]));
  u[i] = uu;
  G[(size_t)n * 384 + 64 + j] = f2bf(r * h[i]);
}

// ---------------------------------------------------------------- h' = u*h + (1-u)*tanh(c)
__global__ void k_hupdate(const float* __restrict__ cand, const float* __restrict__ u,
                          const float* __restrict__ hin, float* __restrict__ hout,
                          u16* __restrict__ houtbf) {
  int i = blockIdx.x * 256 + threadIdx.x;   // < 4096*64
  float c  = tanhf(cand[i]);
  float uu = u[i];
  float h  = uu * hin[i] + (1.0f - uu) * c;
  hout[i] = h;
  if (houtbf) houtbf[i] = f2bf(h);
}

// ---------------------------------------------------------------- prediction head (64 -> 12)
__global__ void k_pred(const float* __restrict__ h, const float* __restrict__ pw,
                       const float* __restrict__ pb, float* __restrict__ out, int tstep) {
  int i = blockIdx.x * 256 + threadIdx.x;   // < 4096*12
  int n = i / 12, p = i - n * 12;
  float v = pb[p];
#pragma unroll
  for (int j = 0; j < 64; j++) v = fmaf(h[(size_t)n * 64 + j], pw[j * 12 + p], v);
  out[(size_t)n * 144 + tstep * 12 + p] = v;
}

// ================================================================ host
extern "C" void kernel_launch(void* const* d_in, const int* in_sizes, int n_in,
                              void* d_out, int out_size, void* d_ws, size_t ws_size,
                              hipStream_t stream) {
  const float* inputs   = (const float*)d_in[0];
  const float* conv1_w  = (const float*)d_in[2];
  const float* conv1_b  = (const float*)d_in[3];
  const float* conv2_w  = (const float*)d_in[4];
  const float* conv2_b  = (const float*)d_in[5];
  const float* bn1_g    = (const float*)d_in[6];
  const float* bn1_b    = (const float*)d_in[7];
  const float* bn2_g    = (const float*)d_in[8];
  const float* bn2_b    = (const float*)d_in[9];
  const float* fcconv_w = (const float*)d_in[10];
  const float* fcconv_b = (const float*)d_in[11];
  const float* fc_w     = (const float*)d_in[12];
  const float* fc_b     = (const float*)d_in[13];
  const float* enc_Wg   = (const float*)d_in[14];
  const float* enc_bg   = (const float*)d_in[15];
  const float* enc_Wc   = (const float*)d_in[16];
  const float* enc_bc   = (const float*)d_in[17];
  const float* dec_Wg   = (const float*)d_in[18];
  const float* dec_bg   = (const float*)d_in[19];
  const float* dec_Wc   = (const float*)d_in[20];
  const float* dec_bc   = (const float*)d_in[21];
  const float* pred_w   = (const float*)d_in[22];
  const float* pred_b   = (const float*)d_in[23];
  const float* adj      = (const float*)d_in[24];
  float* out = (float*)d_out;
  (void)in_sizes; (void)n_in; (void)out_size; (void)ws_size;

  char* wsp = (char*)d_ws;
  auto carve = [&](size_t bytes) -> char* {
    char* p = wsp; wsp += (bytes + 255) & ~(size_t)255; return p;
  };
  u16*   adjbf = (u16*)  carve((size_t)4096 * 4096 * 2);       // 33.5 MB, L2-resident
  u16*   wgE   = (u16*)  carve((size_t)384 * 128 * 2);
  u16*   wcE   = (u16*)  carve((size_t)384 * 64 * 2);
  u16*   wgD   = (u16*)  carve((size_t)384 * 128 * 2);
  u16*   wcD   = (u16*)  carve((size_t)384 * 64 * 2);
  u16*   w2t   = (u16*)  carve((size_t)320 * 64 * 2);          // conv2 weights (K x N)
  float* out1  = (float*)carve((size_t)4096 * 32 * 140 * 4);   // conv1 relu out
  u16*   P     = (u16*)  carve((size_t)270336 * 320 * 2);      // im2col patches (bf16)
  float* Z     = (float*)carve((size_t)270336 * 64 * 4);       // conv2 relu out [m][c2]
  float* stats = (float*)carve(192 * 4);                        // [sum1|sq1|sum2|sq2]
  float* bnp   = (float*)carve(64 * 4);                         // bn1 scale/shift
  u16*   embs  = (u16*)  carve((size_t)24 * 4096 * 64 * 2);
  u16*   G     = (u16*)  carve((size_t)4096 * 384 * 2);         // [X | X1 | X2] bf16
  float* ru    = (float*)carve((size_t)4096 * 128 * 4);
  float* cand  = (float*)carve((size_t)4096 * 64 * 4);
  float* hbuf  = (float*)carve((size_t)4096 * 64 * 4);
  u16*   hbf   = (u16*)  carve((size_t)4096 * 64 * 2);
  float* ubuf  = (float*)carve((size_t)4096 * 64 * 4);
  float* dech  = (float*)carve((size_t)4096 * 64 * 4);

  auto gemm = [&](const u16* A, int lda, const u16* B, int ldb, int M, int N, int K,
                  float* Cf, u16* Cbf, int ldc, const float* bias,
                  const u16* Xsub, int ldx, float alpha, int relu) {
    dim3 g(M / 128, N / 64);
    k_gemm_bf16<<<g, 256, 0, stream>>>(A, lda, B, ldb, M, N, K, Cf, Cbf, ldc,
                                       bias, Xsub, ldx, alpha, relu);
  };

  // ---- one-time conversions + h0 = 0
  k_f2bf<<<65536, 256, 0, stream>>>(adj, adjbf, 4096 * 4096);
  k_f2bf<<<192, 256, 0, stream>>>(enc_Wg, wgE, 384 * 128);
  k_f2bf<<<96,  256, 0, stream>>>(enc_Wc, wcE, 384 * 64);
  k_f2bf<<<192, 256, 0, stream>>>(dec_Wg, wgD, 384 * 128);
  k_f2bf<<<96,  256, 0, stream>>>(dec_Wc, wcD, 384 * 64);
  k_w2t<<<80, 256, 0, stream>>>(conv2_w, w2t);
  k_zero<<<1024, 256, 0, stream>>>((u32*)hbuf, 4096 * 64);
  k_zero<<<512,  256, 0, stream>>>((u32*)hbf,  4096 * 64 / 2);

  // ---- embedding: 24 window steps (conv2 runs on the WMMA GEMM)
  for (int t = 0; t < 24; t++) {
    k_zero<<<1, 256, 0, stream>>>((u32*)stats, 192);
    k_conv1<<<dim3(32, 2240), 256, 0, stream>>>(inputs, conv1_w, conv1_b, out1, stats, t);
    k_bnparams<<<1, 32, 0, stream>>>(stats, bn1_g, bn1_b, bnp);
    k_im2col<<<337920, 256, 0, stream>>>(out1, bnp, P);
    gemm(P, 320, w2t, 64, 270336, 64, 320, Z, nullptr, 64, conv2_b, nullptr, 0, 1.0f, 1);
    k_stats2<<<dim3(64, 1056), 256, 0, stream>>>(Z, stats);
    k_embed_tail<<<4096, 64, 0, stream>>>(Z, stats, bn2_g, bn2_b, fcconv_w, fcconv_b,
                                          fc_w, fc_b, embs, t);
  }

  // ---- one DCGRU cell (two gconvs, each = 2 adjacency WMMA GEMMs + 1 weight GEMM)
  auto cell = [&](int t, const u16* Wg, const float* bg, const u16* Wc, const float* bc,
                  const float* hin, float* hout, u16* houtbf) {
    k_pack<<<1024, 256, 0, stream>>>(embs + (size_t)t * 4096 * 64, hbf, G);
    // gate gconv
    gemm(adjbf, 4096, G,       384, 4096, 128, 4096, nullptr, G + 128, 384, nullptr, nullptr, 0, 1.0f, 0);
    gemm(adjbf, 4096, G + 128, 384, 4096, 128, 4096, nullptr, G + 256, 384, nullptr, G, 384, 2.0f, 0);
    gemm(G, 384, Wg, 128, 4096, 128, 384, ru, nullptr, 128, bg, nullptr, 0, 1.0f, 0);
    k_gate<<<1024, 256, 0, stream>>>(ru, hin, ubuf, G);
    // candidate gconv on [x | r*h]
    gemm(adjbf, 4096, G,       384, 4096, 128, 4096, nullptr, G + 128, 384, nullptr, nullptr, 0, 1.0f, 0);
    gemm(adjbf, 4096, G + 128, 384, 4096, 128, 4096, nullptr, G + 256, 384, nullptr, G, 384, 2.0f, 0);
    gemm(G, 384, Wc, 64, 4096, 64, 384, cand, nullptr, 64, bc, nullptr, 0, 1.0f, 0);
    k_hupdate<<<1024, 256, 0, stream>>>(cand, ubuf, hin, hout, houtbf);
  };

  // ---- encoder (recurrent, h updated in place)
  for (int t = 0; t < 12; t++)
    cell(t, wgE, enc_bg, wcE, enc_bc, hbuf, hbuf, hbf);

  // ---- decoder (12 independent cells from h_enc) + prediction head
  for (int t = 0; t < 12; t++) {
    cell(12 + t, wgD, dec_bg, wcD, dec_bc, hbuf, dech, nullptr);
    k_pred<<<192, 256, 0, stream>>>(dech, pred_w, pred_b, out, t);
  }
}
